// Model_22557168239194
// MI455X (gfx1250) — matrix operand order repaired and checked by reading, not compiled, not run on hardware
//
#include <hip/hip_runtime.h>
#include <hip/hip_bf16.h>
#include <math.h>

typedef _Float16 f16;
typedef __attribute__((ext_vector_type(16))) _Float16 v16h;
typedef __attribute__((ext_vector_type(8)))  _Float16 v8h;
typedef __attribute__((ext_vector_type(8)))  float    v8f;
typedef __attribute__((ext_vector_type(4)))  int      v4i;

#define BB   64
#define TT   256
#define FF   2048
#define CC   512
#define ICC  256
#define BT   16384   // BB*TT
#define KCHUNK 4     // K-tiles staged per sync round (4 * 1KB = 4KB)

#if defined(__HIP_DEVICE_COMPILE__) && __has_builtin(__builtin_amdgcn_global_load_async_to_lds_b128)
#define HAVE_ASYNC_LDS 1
typedef __attribute__((address_space(1))) v4i* gv4i_p;   // global int4*
typedef __attribute__((address_space(3))) v4i* lv4i_p;   // LDS int4*
#else
#define HAVE_ASYNC_LDS 0
#endif

// ---- d_out flat offsets (reference tuple order) ----
#define OFF_SCORE_ABN 0LL
#define OFF_SCORE_NOR 32LL
#define OFF_SEL_NOR   64LL
#define OFF_SEL_ABN1  49216LL
#define OFF_SEL_ABN2  98368LL
#define OFF_SEL_ABN3  147520LL
#define OFF_SCORES    196672LL
#define OFF_SEL_ABN4  213056LL
#define OFF_SEL_ABN5  262208LL
#define OFF_MAG       311360LL
#define OFF_FEATURES  327744LL

// ---------------- zero fill f16 ----------------
__global__ void fill_zero_f16(f16* __restrict__ dst, long long n) {
    long long i = (long long)blockIdx.x * blockDim.x + threadIdx.x;
    long long stride = (long long)gridDim.x * blockDim.x;
    for (; i < n; i += stride) dst[i] = (f16)0.0f;
}

// ---------------- cast f32 -> f16 with per-batch halo padding ----------------
__global__ void cast_pad_f16(const float* __restrict__ src, f16* __restrict__ dst,
                             int Tlen, int cols, int halo, long long total) {
    long long i = (long long)blockIdx.x * blockDim.x + threadIdx.x;
    long long stride = (long long)gridDim.x * blockDim.x;
    const int Tpad = Tlen + 2 * halo;
    for (; i < total; i += stride) {
        long long r = i / cols;
        int f = (int)(i - r * cols);
        int b = (int)(r / Tlen);
        int t = (int)(r - (long long)b * Tlen);
        dst[((long long)b * Tpad + halo + t) * cols + f] = (f16)src[i];
    }
}

// ---------------- pack weights w[o][k][tap] -> WMMA B fragments ----------------
__global__ void pack_w(const float* __restrict__ w, f16* __restrict__ dst,
                       int O, int Kdim, int taps, long long total) {
    long long idx = (long long)blockIdx.x * blockDim.x + threadIdx.x;
    long long stride = (long long)gridDim.x * blockDim.x;
    int ntiles = O >> 4;
    int ktiles = Kdim >> 5;
    for (; idx < total; idx += stride) {
        int j    = (int)(idx & 15);
        int lane = (int)((idx >> 4) & 31);
        long long tile = idx >> 9;
        int nt  = (int)(tile % ntiles);
        long long r = tile / ntiles;
        int kt  = (int)(r % ktiles);
        int tap = (int)(r / ktiles);
        int k = kt * 32 + ((lane >> 4) << 4) + j;
        int n = nt * 16 + (lane & 15);
        dst[idx] = (f16)w[((long long)n * Kdim + k) * taps + tap];
    }
}

// ---------------- pack activation B (per batch), optional transpose ----------------
__global__ void pack_actB(const f16* __restrict__ src, f16* __restrict__ dst,
                          int Kdim, int N, long long srcBS, long long dstBS,
                          int transpose, long long total) {
    long long idx = (long long)blockIdx.x * blockDim.x + threadIdx.x;
    long long stride = (long long)gridDim.x * blockDim.x;
    long long perBatch = (long long)Kdim * N;
    int ntiles = N >> 4;
    for (; idx < total; idx += stride) {
        long long z = idx / perBatch;
        long long e = idx % perBatch;
        int j    = (int)(e & 15);
        int lane = (int)((e >> 4) & 31);
        long long tile = e >> 9;
        int nt = (int)(tile % ntiles);
        int kt = (int)(tile / ntiles);
        int k = kt * 32 + ((lane >> 4) << 4) + j;
        int n = nt * 16 + (lane & 15);
        f16 v = transpose ? src[z * srcBS + (long long)n * Kdim + k]
                          : src[z * srcBS + (long long)k * N + n];
        dst[z * dstBS + e] = v;
    }
}

// ---------------- unified WMMA GEMM with LDS-staged A (async double-buffered) ----------------
__global__ void gemm_wmma(
    const f16* __restrict__ A, long long aBS, int lda, int aColOff, int aHalo,
    const f16* __restrict__ Bp, long long bBS,
    int N, int Kdim, int taps, int dilation, int Tlen,
    const float* __restrict__ bias,
    const float* __restrict__ bnG, const float* __restrict__ bnB,
    int doRelu, float scaleAfter,
    const float* __restrict__ addF32, int addF32Ld,
    const f16* __restrict__ addF16, int addF16Ld,
    float* __restrict__ outF32, long long oF32BS, int ldF32,
    f16* __restrict__ outF16, long long oF16BS, int ldF16, int oF16ColOff, int oF16Halo)
{
    __shared__ __align__(16) f16 smem[2 * KCHUNK * 512];   // 8 KB double buffer

    const int tid  = threadIdx.x;
    const int lane = tid & 31;
    const int wave = tid >> 5;
    const int rt = blockIdx.x;
    const int ct = blockIdx.y * 8 + wave;
    const int z  = blockIdx.z;
    const bool doC = (ct * 16 < N);

    const int kTiles = Kdim >> 5;
    const int nTiles = N >> 4;
    const int ktot = taps * kTiles;
    const int nChunks = ktot / KCHUNK;

    const f16* Ab = A + (long long)z * aBS;
    const f16* Bb = Bp + (long long)z * bBS;

    // ---- staging mapping: 256 threads move one b128 each -> 4KB chunk ----
    const int sm_ = (tid >> 2) & 15;   // row within 16-row tile
    const int seg = tid & 3;           // 8-half segment within 32-half row
    const int tc0 = tid >> 6;          // which of the 4 K-tiles in the chunk
    const int srow = rt * 16 + sm_;
    const int sb = srow / Tlen;
    const int st = srow - sb * Tlen;
    const int Tpad = Tlen + 2 * aHalo;

    // ---- consumer mapping (contiguous k = 16*hi + s, same order as packed B) ----
    const int mIn = lane & 15;
    const int hi  = lane >> 4;
    const f16* Bbase = Bb + (long long)ct * 512 + lane * 16;

    v8f acc;
#pragma unroll
    for (int i = 0; i < 8; ++i) acc[i] = 0.0f;

    auto stageChunk = [&](int chunk, int bufSel) {
        const int flat = chunk * KCHUNK + tc0;
        const int tap = flat / kTiles;
        const int kt  = flat - tap * kTiles;
        const int dt  = (tap - (taps >> 1)) * dilation;
        const long long prow = aHalo ? ((long long)sb * Tpad + aHalo + st + dt)
                                     : (long long)(srow + dt);
        const f16* g = Ab + prow * lda + aColOff + kt * 32 + seg * 8;
        f16* l = smem + bufSel * (KCHUNK * 512) + tc0 * 512 + sm_ * 32 + seg * 8;
#if HAVE_ASYNC_LDS
        __builtin_amdgcn_global_load_async_to_lds_b128(
            (gv4i_p)(unsigned long long)(const void*)g,
            (lv4i_p)(unsigned int)(unsigned long long)(void*)l,
            0, 0);
#else
        *(v8h*)l = *(const v8h*)g;
#endif
    };

    stageChunk(0, 0);
    for (int c = 0; c < nChunks; ++c) {
        const int buf = c & 1;
#if HAVE_ASYNC_LDS
#if __has_builtin(__builtin_amdgcn_s_wait_asynccnt)
        __builtin_amdgcn_s_wait_asynccnt(0);
#else
        asm volatile("s_wait_asynccnt 0" ::: "memory");
#endif
#endif
        __syncthreads();
        if (c + 1 < nChunks) stageChunk(c + 1, buf ^ 1);
        if (doC) {
            const f16* lb = smem + buf * (KCHUNK * 512);
#pragma unroll
            for (int tcc = 0; tcc < KCHUNK; ++tcc) {
                const int flat = c * KCHUNK + tcc;
                const f16* lr = lb + tcc * 512 + mIn * 32 + hi * 16;
                v8h a0 = *(const v8h*)(lr);
                v8h a1 = *(const v8h*)(lr + 8);
                v16h a = __builtin_shufflevector(a0, a1,
                         0,1,2,3,4,5,6,7,8,9,10,11,12,13,14,15);
                const f16* bp = Bbase + (long long)flat * nTiles * 512;
                v16h b = *(const v16h*)bp;
                __builtin_prefetch((const void*)(bp + (long long)KCHUNK * nTiles * 512), 0, 1);
                acc = __builtin_amdgcn_wmma_f32_16x16x32_f16(false, a, false, b,
                                                             (short)0, acc, false, false);
            }
        }
    }

    if (!doC) return;

    // -------- epilogue --------
    const int col = ct * 16 + mIn;
    const float bv = bias ? bias[col] : 0.0f;
    float gsc = 1.0f, gbt = 0.0f;
    if (bnG) { gsc = bnG[col] * rsqrtf(1.0f + 1e-5f); gbt = bnB[col]; }
    const int mo = hi * 8;
#pragma unroll
    for (int v = 0; v < 8; ++v) {
        const int row = rt * 16 + mo + v;
        float val = acc[v] * scaleAfter + bv;
        if (doRelu) val = fmaxf(val, 0.0f);
        if (bnG) val = val * gsc + gbt;
        if (addF32) val += addF32[(long long)row * addF32Ld + col];
        if (addF16) val += (float)addF16[(long long)row * addF16Ld + col];
        if (outF32) outF32[(long long)z * oF32BS + (long long)row * ldF32 + col] = val;
        if (outF16) {
            long long orow = row;
            if (oF16Halo) {
                int rb = row / Tlen, rtt = row - rb * Tlen;
                orow = (long long)rb * (Tlen + 2 * oF16Halo) + oF16Halo + rtt;
            }
            outF16[(long long)z * oF16BS + orow * ldF16 + oF16ColOff + col] = (f16)val;
        }
    }
}

// ---------------- fc3 (128 -> 1) + sigmoid ----------------
__global__ void fc3_sigmoid(const f16* __restrict__ h2, const float* __restrict__ w,
                            const float* __restrict__ b, float* __restrict__ scoresOut, int rows) {
    int r = blockIdx.x * blockDim.x + threadIdx.x;
    if (r >= rows) return;
    float s = b[0];
#pragma unroll 4
    for (int i = 0; i < 128; ++i) s += (float)h2[(long long)r * 128 + i] * w[i];
    scoresOut[r] = 1.0f / (1.0f + expf(-s));
}

// ---------------- feature magnitude (L2 over 512) ----------------
__global__ void feat_mag_k(const float* __restrict__ features, float* __restrict__ magOut, int rows) {
    int r = blockIdx.x * blockDim.x + threadIdx.x;
    if (r >= rows) return;
    float s = 0.0f;
#pragma unroll 4
    for (int i = 0; i < 512; ++i) {
        float v = features[(long long)r * 512 + i];
        s += v * v;
    }
    magOut[r] = sqrtf(s);
}

// ---------------- top-k(3) head: select + means ----------------
__global__ void head_topk(const float* __restrict__ mag, const float* __restrict__ scores,
                          const float* __restrict__ features, float* __restrict__ out) {
    __shared__ float sm[256];
    __shared__ int   sidx[3];
    __shared__ float smean;
    const int b = blockIdx.x;
    const int t = threadIdx.x;
    sm[t] = mag[b * 256 + t];
    __syncthreads();
    if (t == 0) {
        int ids[3];
        for (int j = 0; j < 3; ++j) {
            int bi = 0; float bvv = -1e30f;
            for (int i = 0; i < 256; ++i) { float v = sm[i]; if (v > bvv) { bvv = v; bi = i; } }
            ids[j] = bi; sm[bi] = -1e30f;
        }
        float ssum = 0.0f;
        for (int j = 0; j < 3; ++j) ssum += scores[b * 256 + ids[j]];
        smean = ssum * (1.0f / 3.0f);
        sidx[0] = ids[0]; sidx[1] = ids[1]; sidx[2] = ids[2];
    }
    __syncthreads();
    const bool abn = (b >= 32);
    const int hb = abn ? (b - 32) : b;
    for (int j = 0; j < 3; ++j) {
        const long long srcRow = (long long)b * 256 + sidx[j];
        for (int c = t; c < 512; c += 256) {
            float v = features[srcRow * 512 + c];
            long long o = ((long long)(hb * 3 + j)) * 512 + c;
            if (abn) {
                out[OFF_SEL_ABN1 + o] = v;
                out[OFF_SEL_ABN2 + o] = v;
                out[OFF_SEL_ABN3 + o] = v;
                out[OFF_SEL_ABN4 + o] = v;
                out[OFF_SEL_ABN5 + o] = v;
            } else {
                out[OFF_SEL_NOR + o] = v;
            }
        }
    }
    if (t == 0) {
        if (abn) out[OFF_SCORE_ABN + hb] = smean;
        else     out[OFF_SCORE_NOR + hb] = smean;
    }
}

// =====================================================================
extern "C" void kernel_launch(void* const* d_in, const int* in_sizes, int n_in,
                              void* d_out, int out_size, void* d_ws, size_t ws_size,
                              hipStream_t stream) {
    (void)in_sizes; (void)n_in; (void)out_size; (void)ws_size;
    const float* in_x   = (const float*)d_in[0];
    const float* c1_w   = (const float*)d_in[1];
    const float* c1_b   = (const float*)d_in[2];
    const float* bn1_g  = (const float*)d_in[3];
    const float* bn1_b  = (const float*)d_in[4];
    const float* c2_w   = (const float*)d_in[5];
    const float* c2_b   = (const float*)d_in[6];
    const float* bn2_g  = (const float*)d_in[7];
    const float* bn2_b  = (const float*)d_in[8];
    const float* c3_w   = (const float*)d_in[9];
    const float* c3_b   = (const float*)d_in[10];
    const float* bn3_g  = (const float*)d_in[11];
    const float* bn3_b  = (const float*)d_in[12];
    const float* c4_w   = (const float*)d_in[13];
    const float* g_w    = (const float*)d_in[14];
    const float* g_b    = (const float*)d_in[15];
    const float* th_w   = (const float*)d_in[16];
    const float* th_b   = (const float*)d_in[17];
    const float* ph_w   = (const float*)d_in[18];
    const float* ph_b   = (const float*)d_in[19];
    const float* W_w    = (const float*)d_in[20];
    const float* W_b    = (const float*)d_in[21];
    const float* Wbn_g  = (const float*)d_in[22];
    const float* Wbn_b  = (const float*)d_in[23];
    const float* c5_w   = (const float*)d_in[24];
    const float* bn5_g  = (const float*)d_in[25];
    const float* bn5_b  = (const float*)d_in[26];
    const float* res_w  = (const float*)d_in[27];
    const float* res_b  = (const float*)d_in[28];
    const float* fc1_w  = (const float*)d_in[29];
    const float* fc1_b  = (const float*)d_in[30];
    const float* fc2_w  = (const float*)d_in[31];
    const float* fc2_b  = (const float*)d_in[32];
    const float* fc3_w  = (const float*)d_in[33];
    const float* fc3_b  = (const float*)d_in[34];

    float* out = (float*)d_out;

    // ---- workspace layout ----
    char* wsb = (char*)d_ws;
    size_t off = 0;
    auto alloc = [&](size_t bytes) -> void* {
        void* p = wsb + off;
        off += (bytes + 255) & ~(size_t)255;
        return p;
    };
    const int XH = 4;
    const long long XPADROWS  = (long long)BB * (TT + 2 * XH);
    const long long CATROWS   = (long long)BB * (TT + 2);
    f16* X16pad = (f16*)alloc((size_t)XPADROWS * FF * 2);
    f16* catpad = (f16*)alloc((size_t)CATROWS * 2048 * 2);
    f16* Wc1p   = (f16*)alloc((size_t)3 * FF * CC * 2);
    f16* Wc2p   = (f16*)alloc((size_t)3 * FF * CC * 2);
    f16* Wc3p   = (f16*)alloc((size_t)3 * FF * CC * 2);
    f16* Wc4p   = (f16*)alloc((size_t)(3 * CC) * CC * 2);
    f16* Wgp    = (f16*)alloc((size_t)CC * ICC * 2);
    f16* Wthp   = (f16*)alloc((size_t)CC * ICC * 2);
    f16* Wphp   = (f16*)alloc((size_t)CC * ICC * 2);
    f16* WWp    = (f16*)alloc((size_t)ICC * CC * 2);
    f16* Wc5p   = (f16*)alloc((size_t)3 * FF * CC * 2);
    f16* Wresp  = (f16*)alloc((size_t)FF * CC * 2);
    f16* Wfc1p  = (f16*)alloc((size_t)512 * 512 * 2);
    f16* Wfc2p  = (f16*)alloc((size_t)512 * 128 * 2);
    f16* outbuf = (f16*)alloc((size_t)BT * CC * 2);
    f16* gx_a   = (f16*)alloc((size_t)BT * ICC * 2);
    f16* th_a   = (f16*)alloc((size_t)BT * ICC * 2);
    f16* ph_a   = (f16*)alloc((size_t)BT * ICC * 2);
    f16* phPack = (f16*)alloc((size_t)BT * ICC * 2);
    f16* gxPack = (f16*)alloc((size_t)BT * ICC * 2);
    f16* fbuf   = (f16*)alloc((size_t)BB * TT * TT * 2);
    f16* y_t    = (f16*)alloc((size_t)BT * ICC * 2);
    float* resbuf = (float*)alloc((size_t)BT * CC * 4);
    f16* feat16 = (f16*)alloc((size_t)BT * CC * 2);
    f16* h1     = (f16*)alloc((size_t)BT * 512 * 2);
    f16* h2     = (f16*)alloc((size_t)BT * 128 * 2);

    const dim3 blk(256);

    fill_zero_f16<<<dim3(16384), blk, 0, stream>>>(X16pad, XPADROWS * FF);
    fill_zero_f16<<<dim3(16384), blk, 0, stream>>>(catpad, CATROWS * 2048);
    cast_pad_f16<<<dim3(16384), blk, 0, stream>>>(in_x, X16pad, TT, FF, XH, (long long)BT * FF);

    pack_w<<<dim3(8192), blk, 0, stream>>>(c1_w,  Wc1p,  CC, FF, 3, (long long)3 * FF * CC);
    pack_w<<<dim3(8192), blk, 0, stream>>>(c2_w,  Wc2p,  CC, FF, 3, (long long)3 * FF * CC);
    pack_w<<<dim3(8192), blk, 0, stream>>>(c3_w,  Wc3p,  CC, FF, 3, (long long)3 * FF * CC);
    pack_w<<<dim3(4096), blk, 0, stream>>>(c4_w,  Wc4p,  CC, 3 * CC, 1, (long long)(3 * CC) * CC);
    pack_w<<<dim3(1024), blk, 0, stream>>>(g_w,   Wgp,   ICC, CC, 1, (long long)CC * ICC);
    pack_w<<<dim3(1024), blk, 0, stream>>>(th_w,  Wthp,  ICC, CC, 1, (long long)CC * ICC);
    pack_w<<<dim3(1024), blk, 0, stream>>>(ph_w,  Wphp,  ICC, CC, 1, (long long)CC * ICC);
    pack_w<<<dim3(1024), blk, 0, stream>>>(W_w,   WWp,   CC, ICC, 1, (long long)ICC * CC);
    pack_w<<<dim3(8192), blk, 0, stream>>>(c5_w,  Wc5p,  CC, FF, 3, (long long)3 * FF * CC);
    pack_w<<<dim3(4096), blk, 0, stream>>>(res_w, Wresp, CC, FF, 1, (long long)FF * CC);
    pack_w<<<dim3(1024), blk, 0, stream>>>(fc1_w, Wfc1p, 512, 512, 1, (long long)512 * 512);
    pack_w<<<dim3(256),  blk, 0, stream>>>(fc2_w, Wfc2p, 128, 512, 1, (long long)512 * 128);

    gemm_wmma<<<dim3(BT/16, CC/128, 1), blk, 0, stream>>>(
        X16pad, 0, FF, 0, XH, Wc1p, 0, CC, FF, 3, 1, TT,
        c1_b, bn1_g, bn1_b, 1, 1.0f, nullptr, 0, nullptr, 0,
        nullptr, 0, 0, catpad, 0, 2048, 512, 1);
    gemm_wmma<<<dim3(BT/16, CC/128, 1), blk, 0, stream>>>(
        X16pad, 0, FF, 0, XH, Wc2p, 0, CC, FF, 3, 2, TT,
        c2_b, bn2_g, bn2_b, 1, 1.0f, nullptr, 0, nullptr, 0,
        nullptr, 0, 0, catpad, 0, 2048, 1024, 1);
    gemm_wmma<<<dim3(BT/16, CC/128, 1), blk, 0, stream>>>(
        X16pad, 0, FF, 0, XH, Wc3p, 0, CC, FF, 3, 4, TT,
        c3_b, bn3_g, bn3_b, 1, 1.0f, nullptr, 0, nullptr, 0,
        nullptr, 0, 0, catpad, 0, 2048, 1536, 1);

    gemm_wmma<<<dim3(BT/16, CC/128, 1), blk, 0, stream>>>(
        catpad, 0, 2048, 512, 1, Wc4p, 0, CC, 3 * CC, 1, 1, TT,
        nullptr, nullptr, nullptr, 1, 1.0f, nullptr, 0, nullptr, 0,
        nullptr, 0, 0, outbuf, 0, CC, 0, 0);

    gemm_wmma<<<dim3(BT/16, ICC/128, 1), blk, 0, stream>>>(
        outbuf, 0, CC, 0, 0, Wgp, 0, ICC, CC, 1, 1, TT,
        g_b, nullptr, nullptr, 0, 1.0f, nullptr, 0, nullptr, 0,
        nullptr, 0, 0, gx_a, 0, ICC, 0, 0);
    gemm_wmma<<<dim3(BT/16, ICC/128, 1), blk, 0, stream>>>(
        outbuf, 0, CC, 0, 0, Wthp, 0, ICC, CC, 1, 1, TT,
        th_b, nullptr, nullptr, 0, 1.0f, nullptr, 0, nullptr, 0,
        nullptr, 0, 0, th_a, 0, ICC, 0, 0);
    gemm_wmma<<<dim3(BT/16, ICC/128, 1), blk, 0, stream>>>(
        outbuf, 0, CC, 0, 0, Wphp, 0, ICC, CC, 1, 1, TT,
        ph_b, nullptr, nullptr, 0, 1.0f, nullptr, 0, nullptr, 0,
        nullptr, 0, 0, ph_a, 0, ICC, 0, 0);

    pack_actB<<<dim3(8192), blk, 0, stream>>>(ph_a, phPack, ICC, TT,
        (long long)TT * ICC, (long long)TT * ICC, 1, (long long)BB * TT * ICC);
    pack_actB<<<dim3(8192), blk, 0, stream>>>(gx_a, gxPack, TT, ICC,
        (long long)TT * ICC, (long long)TT * ICC, 0, (long long)BB * TT * ICC);

    gemm_wmma<<<dim3(TT/16, TT/128, BB), blk, 0, stream>>>(
        th_a, (long long)TT * ICC, ICC, 0, 0, phPack, (long long)TT * ICC,
        TT, ICC, 1, 1, TT,
        nullptr, nullptr, nullptr, 0, 1.0f / (float)TT, nullptr, 0, nullptr, 0,
        nullptr, 0, 0, fbuf, (long long)TT * TT, TT, 0, 0);

    gemm_wmma<<<dim3(TT/16, ICC/128, BB), blk, 0, stream>>>(
        fbuf, (long long)TT * TT, TT, 0, 0, gxPack, (long long)TT * ICC,
        ICC, TT, 1, 1, TT,
        nullptr, nullptr, nullptr, 0, 1.0f, nullptr, 0, nullptr, 0,
        nullptr, 0, 0, y_t, (long long)TT * ICC, ICC, 0, 0);

    gemm_wmma<<<dim3(BT/16, CC/128, 1), blk, 0, stream>>>(
        y_t, 0, ICC, 0, 0, WWp, 0, CC, ICC, 1, 1, TT,
        W_b, Wbn_g, Wbn_b, 0, 1.0f, nullptr, 0, outbuf, CC,
        nullptr, 0, 0, catpad, 0, 2048, 0, 1);

    gemm_wmma<<<dim3(BT/16, CC/128, 1), blk, 0, stream>>>(
        X16pad, 0, FF, 0, XH, Wresp, 0, CC, FF, 1, 1, TT,
        res_b, nullptr, nullptr, 0, 1.0f, nullptr, 0, nullptr, 0,
        resbuf, 0, CC, nullptr, 0, 0, 0, 0);

    gemm_wmma<<<dim3(BT/16, CC/128, 1), blk, 0, stream>>>(
        catpad, 0, 2048, 0, 1, Wc5p, 0, CC, FF, 3, 1, TT,
        nullptr, bn5_g, bn5_b, 1, 1.0f, resbuf, CC, nullptr, 0,
        out + OFF_FEATURES, 0, CC, feat16, 0, CC, 0, 0);

    gemm_wmma<<<dim3(BT/16, 512/128, 1), blk, 0, stream>>>(
        feat16, 0, CC, 0, 0, Wfc1p, 0, 512, 512, 1, 1, TT,
        fc1_b, nullptr, nullptr, 1, 1.0f, nullptr, 0, nullptr, 0,
        nullptr, 0, 0, h1, 0, 512, 0, 0);
    gemm_wmma<<<dim3(BT/16, 1, 1), blk, 0, stream>>>(
        h1, 0, 512, 0, 0, Wfc2p, 0, 128, 512, 1, 1, TT,
        fc2_b, nullptr, nullptr, 1, 1.0f, nullptr, 0, nullptr, 0,
        nullptr, 0, 0, h2, 0, 128, 0, 0);

    fc3_sigmoid<<<dim3(BT/256), blk, 0, stream>>>(h2, fc3_w, fc3_b, out + OFF_SCORES, BT);
    feat_mag_k<<<dim3(BT/256), blk, 0, stream>>>(out + OFF_FEATURES, out + OFF_MAG, BT);
    head_topk<<<dim3(BB), blk, 0, stream>>>(out + OFF_MAG, out + OFF_SCORES,
                                            out + OFF_FEATURES, out);
}
